// TinyViT_25520695673201
// MI455X (gfx1250) — matrix operand order repaired and checked by reading, not compiled, not run on hardware
//
#include <hip/hip_runtime.h>
#include <hip/hip_bf16.h>
#include <math.h>

typedef __attribute__((ext_vector_type(16))) _Float16 v16h;
typedef __attribute__((ext_vector_type(8)))  _Float16 v8h;
typedef __attribute__((ext_vector_type(8)))  float    v8f;

#define LDS_STR 40   // halves per row (80B): keeps 8-half chunks 16B aligned

// ---------------------------------------------------------------------------
// Weight conversion: W[K,N] f32 -> Wt[N,K] f16 (transposed so B-fragments are
// contiguous-K per output column, matching the WMMA B layout)
// ---------------------------------------------------------------------------
__global__ __launch_bounds__(256) void wt_convert(const float* __restrict__ src,
                                                  _Float16* __restrict__ dst,
                                                  int K, int N) {
    size_t idx = (size_t)blockIdx.x * 256 + threadIdx.x;
    if (idx >= (size_t)K * N) return;
    int n = (int)(idx % N);
    int k = (int)(idx / N);
    dst[(size_t)n * K + k] = (_Float16)src[idx];
}

// ---------------------------------------------------------------------------
// LayerNorm over C=384, emit f16 activations for the WMMA GEMMs
// ---------------------------------------------------------------------------
__global__ __launch_bounds__(128) void ln_f16_kernel(const float* __restrict__ X,
                                                     const float* __restrict__ g,
                                                     const float* __restrict__ b,
                                                     _Float16* __restrict__ out,
                                                     int C) {
    __shared__ float red[128];
    int row = blockIdx.x;
    const float* xr = X + (size_t)row * C;
    float s = 0.f, s2 = 0.f;
    for (int c = threadIdx.x; c < C; c += 128) {
        float v = xr[c];
        s += v; s2 += v * v;
    }
    red[threadIdx.x] = s; __syncthreads();
    for (int off = 64; off > 0; off >>= 1) {
        if (threadIdx.x < off) red[threadIdx.x] += red[threadIdx.x + off];
        __syncthreads();
    }
    float mean = red[0] / (float)C;
    __syncthreads();
    red[threadIdx.x] = s2; __syncthreads();
    for (int off = 64; off > 0; off >>= 1) {
        if (threadIdx.x < off) red[threadIdx.x] += red[threadIdx.x + off];
        __syncthreads();
    }
    float var = red[0] / (float)C - mean * mean;
    float inv = rsqrtf(var + 1e-5f);
    for (int c = threadIdx.x; c < C; c += 128) {
        float v = (xr[c] - mean) * inv * g[c] + b[c];
        out[(size_t)row * C + c] = (_Float16)v;
    }
}

// ---------------------------------------------------------------------------
// WMMA GEMM: C[M,N] = A[M,K](f16) @ Wt[N,K]^T(f16) + bias, with epilogues.
// Block 256 (8 waves, 4M x 2N), block tile 128x128, wave tile 32x64
// (2x4 WMMA 16x16 tiles -> 8 wmma per 12 ds_load_b128 per K-step).
// Double-buffered LDS: stage tile k+1 while computing tile k, one barrier
// per K-step. EPI: 0=bias->f32, 1=bias+residual->f32, 2=bias+GELU->f16.
// ---------------------------------------------------------------------------
template <int EPI>
__global__ __launch_bounds__(256) void gemm_wmma(const _Float16* __restrict__ A,
                                                 const _Float16* __restrict__ Bt,
                                                 const float* __restrict__ bias,
                                                 const float* __restrict__ Res,
                                                 void* __restrict__ Cout,
                                                 int M, int N, int K) {
    __shared__ _Float16 As[2][128 * LDS_STR];
    __shared__ _Float16 Bs[2][128 * LDS_STR];

    const int tid   = threadIdx.x;
    const int lane  = tid & 31;
    const int wave  = tid >> 5;
    const int waveM = wave & 3;   // 0..3 -> 32-row slabs
    const int waveN = wave >> 2;  // 0..1 -> 64-col slabs

    const int mBlock = blockIdx.x * 128;
    const int nBlock = blockIdx.y * 128;

    const int  laneHalf = lane & 15;
    const bool hiLane   = lane >= 16;

    // staging coordinates (A tile 128x32 halves, B tile 128x32 halves)
    const int sRow = tid >> 2;
    const int sCol = (tid & 3) * 8;

    v8f acc[2][4] = {};

    auto stage = [&](int buf, int k0) {
#pragma unroll
        for (int i = 0; i < 2; ++i) {
            int row = sRow + i * 64;
            *(v8h*)&As[buf][row * LDS_STR + sCol] =
                *(const v8h*)&A[(size_t)(mBlock + row) * K + k0 + sCol];
        }
#pragma unroll
        for (int i = 0; i < 2; ++i) {
            int row = sRow + i * 64;
            *(v8h*)&Bs[buf][row * LDS_STR + sCol] =
                *(const v8h*)&Bt[(size_t)(nBlock + row) * K + k0 + sCol];
        }
    };

    stage(0, 0);
    __syncthreads();
    int buf = 0;

    for (int k0 = 0; k0 < K; k0 += 32) {
        if (k0 + 32 < K) {
            stage(buf ^ 1, k0 + 32);  // overlap next-tile staging with WMMAs
            if (k0 + 64 < K) {        // prefetch tile after next -> global_prefetch_b8
                __builtin_prefetch(&A[(size_t)(mBlock + sRow) * K + k0 + 64], 0, 1);
                __builtin_prefetch(&Bt[(size_t)(nBlock + sRow) * K + k0 + 64], 0, 1);
            }
        }

        // A fragments per ISA layout: lanes0-15 M=row, K 0..7 & 16..23;
        //                             lanes16-31 M=row, K 8..15 & 24..31
        v16h afrag[2], bfrag[4];
#pragma unroll
        for (int ti = 0; ti < 2; ++ti) {
            int row = waveM * 32 + ti * 16 + laneHalf;
            int kb  = hiLane ? 8 : 0;
            v8h lo = *(const v8h*)&As[buf][row * LDS_STR + kb];
            v8h hi = *(const v8h*)&As[buf][row * LDS_STR + kb + 16];
            afrag[ti] = __builtin_shufflevector(lo, hi, 0, 1, 2, 3, 4, 5, 6, 7,
                                                8, 9, 10, 11, 12, 13, 14, 15);
        }
        // B fragments: lanes0-15 N=col, K 0..7 & 16..23; lanes16-31 N=col, K 8..15 & 24..31
#pragma unroll
        for (int tj = 0; tj < 4; ++tj) {
            int row = waveN * 64 + tj * 16 + laneHalf;
            int kb  = hiLane ? 8 : 0;
            v8h lo = *(const v8h*)&Bs[buf][row * LDS_STR + kb];
            v8h hi = *(const v8h*)&Bs[buf][row * LDS_STR + kb + 16];
            bfrag[tj] = __builtin_shufflevector(lo, hi, 0, 1, 2, 3, 4, 5, 6, 7,
                                                8, 9, 10, 11, 12, 13, 14, 15);
        }
#pragma unroll
        for (int ti = 0; ti < 2; ++ti)
#pragma unroll
            for (int tj = 0; tj < 4; ++tj)
                acc[ti][tj] = __builtin_amdgcn_wmma_f32_16x16x32_f16(
                    false, afrag[ti], false, bfrag[tj], (short)0, acc[ti][tj],
                    false, false);

        __syncthreads();
        buf ^= 1;
    }

    // epilogue: D layout VGPR r -> M = r (+8 for lanes 16..31), N = lane&15
#pragma unroll
    for (int ti = 0; ti < 2; ++ti) {
#pragma unroll
        for (int tj = 0; tj < 4; ++tj) {
            int n = nBlock + waveN * 64 + tj * 16 + laneHalf;
            float bv = bias[n];
#pragma unroll
            for (int r = 0; r < 8; ++r) {
                int m = mBlock + waveM * 32 + ti * 16 + r + (hiLane ? 8 : 0);
                size_t idx = (size_t)m * N + n;
                float v = acc[ti][tj][r] + bv;
                if (EPI == 1) v += Res[idx];
                if (EPI == 2) {
                    v = 0.5f * v * (1.0f + erff(v * 0.70710678118654752f));
                    ((_Float16*)Cout)[idx] = (_Float16)v;
                } else {
                    ((float*)Cout)[idx] = v;
                }
            }
        }
    }
}

// ---------------------------------------------------------------------------
// Windowed attention: one 64-thread block per (window, head). N=49 tokens,
// d=32. k/v staged in LDS; softmax per query row in registers; output written
// straight back to token order (window permutation folded into indexing).
// ---------------------------------------------------------------------------
__global__ __launch_bounds__(64) void attn_kernel(const float* __restrict__ qkv,
                                                  const float* __restrict__ biases,
                                                  const int* __restrict__ bidx,
                                                  _Float16* __restrict__ out,
                                                  int noff) {
    __shared__ float ks[49 * 32];
    __shared__ float vs[49 * 32];
    __shared__ int   rowIdx[49];

    int blk  = blockIdx.x;
    int head = blk % 12;
    int wi   = blk / 12;
    int b    = wi >> 6;           // 64 windows per image (8x8)
    int r    = wi & 63;
    int wh   = r >> 3, ww = r & 7;

    if (threadIdx.x < 49) {
        int n  = threadIdx.x;
        int gh = wh * 7 + n / 7;
        int gw = ww * 7 + n % 7;
        rowIdx[n] = b * 3136 + gh * 56 + gw;
    }
    __syncthreads();

    for (int i = threadIdx.x; i < 49 * 32; i += 64) {
        int n = i >> 5, d = i & 31;
        size_t base = (size_t)rowIdx[n] * 1152 + head * 96;
        ks[i] = qkv[base + 32 + d];
        vs[i] = qkv[base + 64 + d];
    }
    __syncthreads();

    int n = threadIdx.x;
    if (n < 49) {
        float q[32];
        size_t base = (size_t)rowIdx[n] * 1152 + head * 96;
#pragma unroll
        for (int d = 0; d < 32; ++d) q[d] = qkv[base + d];

        float logits[49], mx = -1e30f;
        for (int m = 0; m < 49; ++m) {
            float s = 0.f;
#pragma unroll
            for (int d = 0; d < 32; ++d) s += q[d] * ks[m * 32 + d];
            s = s * 0.17677669529663687f + biases[head * noff + bidx[n * 49 + m]];
            logits[m] = s;
            mx = fmaxf(mx, s);
        }
        float den = 0.f;
        for (int m = 0; m < 49; ++m) {
            logits[m] = __expf(logits[m] - mx);
            den += logits[m];
        }
        float rden = 1.f / den;
        float acc[32];
#pragma unroll
        for (int d = 0; d < 32; ++d) acc[d] = 0.f;
        for (int m = 0; m < 49; ++m) {
            float w = logits[m] * rden;
#pragma unroll
            for (int d = 0; d < 32; ++d) acc[d] += w * vs[m * 32 + d];
        }
        _Float16* op = out + (size_t)rowIdx[n] * 384 + head * 32;
#pragma unroll
        for (int d = 0; d < 32; ++d) op[d] = (_Float16)acc[d];
    }
}

// ---------------------------------------------------------------------------
// Depthwise 3x3 conv (NHWC, zero pad) + BatchNorm. Pure bandwidth.
// ---------------------------------------------------------------------------
__global__ __launch_bounds__(256) void dwconv_bn(const float* __restrict__ X,
                                                 const float* __restrict__ W,
                                                 const float* __restrict__ bnw,
                                                 const float* __restrict__ bnb,
                                                 const float* __restrict__ bnm,
                                                 const float* __restrict__ bnv,
                                                 float* __restrict__ Y) {
    size_t idx = (size_t)blockIdx.x * 256 + threadIdx.x;  // over B*H*W*C
    int c = (int)(idx % 384);
    size_t p = idx / 384;
    int w = (int)(p % 56);
    size_t t = p / 56;
    int h = (int)(t % 56);
    int b = (int)(t / 56);
    float s = 0.f;
#pragma unroll
    for (int kh = 0; kh < 3; ++kh) {
        int hh = h + kh - 1;
        if (hh < 0 || hh >= 56) continue;
#pragma unroll
        for (int kw = 0; kw < 3; ++kw) {
            int wp = w + kw - 1;
            if (wp < 0 || wp >= 56) continue;
            s += X[((size_t)b * 3136 + hh * 56 + wp) * 384 + c] * W[c * 9 + kh * 3 + kw];
        }
    }
    float inv = bnw[c] * rsqrtf(bnv[c] + 1e-5f);
    Y[idx] = (s - bnm[c]) * inv + bnb[c];
}

// ---------------------------------------------------------------------------
extern "C" void kernel_launch(void* const* d_in, const int* in_sizes, int n_in,
                              void* d_out, int out_size, void* d_ws, size_t ws_size,
                              hipStream_t stream) {
    const float* x       = (const float*)d_in[0];
    const float* ln1_g   = (const float*)d_in[1];
    const float* ln1_b   = (const float*)d_in[2];
    const float* qkv_w   = (const float*)d_in[3];
    const float* qkv_b   = (const float*)d_in[4];
    const float* abias   = (const float*)d_in[5];
    const int*   bidx    = (const int*)d_in[6];
    const float* proj_w  = (const float*)d_in[7];
    const float* proj_b  = (const float*)d_in[8];
    const float* conv_w  = (const float*)d_in[9];
    const float* bn_w    = (const float*)d_in[10];
    const float* bn_b    = (const float*)d_in[11];
    const float* bn_mean = (const float*)d_in[12];
    const float* bn_var  = (const float*)d_in[13];
    const float* ln2_g   = (const float*)d_in[14];
    const float* ln2_b   = (const float*)d_in[15];
    const float* fc1_w   = (const float*)d_in[16];
    const float* fc1_b   = (const float*)d_in[17];
    const float* fc2_w   = (const float*)d_in[18];
    const float* fc2_b   = (const float*)d_in[19];

    const int M = 32 * 56 * 56;  // 100352 tokens
    const int C = 384, QN = 1152, HID = 1536;
    int noff = in_sizes[5] / 12;

    char* ws = (char*)d_ws;
    size_t off = 0;
    auto alloc = [&](size_t bytes) {
        size_t o = off;
        off += (bytes + 255) & ~(size_t)255;
        return o;
    };
    _Float16* wt_qkv  = (_Float16*)(ws + alloc((size_t)C * QN * 2));
    _Float16* wt_proj = (_Float16*)(ws + alloc((size_t)C * C * 2));
    _Float16* wt_fc1  = (_Float16*)(ws + alloc((size_t)C * HID * 2));
    _Float16* wt_fc2  = (_Float16*)(ws + alloc((size_t)HID * C * 2));
    _Float16* A1      = (_Float16*)(ws + alloc((size_t)M * C * 2));
    float*    qkv     = (float*)   (ws + alloc((size_t)M * QN * 4));
    _Float16* attn_o  = (_Float16*)(ws + alloc((size_t)M * C * 2));
    float*    x1      = (float*)   (ws + alloc((size_t)M * C * 4));
    float*    x2      = (float*)   (ws + alloc((size_t)M * C * 4));
    _Float16* A3      = (_Float16*)(ws + alloc((size_t)M * C * 2));
    _Float16* hmid    = (_Float16*)(ws + alloc((size_t)M * HID * 2));

    // one-time weight transposes to f16
    wt_convert<<<(C * QN + 255) / 256, 256, 0, stream>>>(qkv_w, wt_qkv, C, QN);
    wt_convert<<<(C * C + 255) / 256, 256, 0, stream>>>(proj_w, wt_proj, C, C);
    wt_convert<<<(C * HID + 255) / 256, 256, 0, stream>>>(fc1_w, wt_fc1, C, HID);
    wt_convert<<<(HID * C + 255) / 256, 256, 0, stream>>>(fc2_w, wt_fc2, HID, C);

    // LN1 -> f16
    ln_f16_kernel<<<M, 128, 0, stream>>>(x, ln1_g, ln1_b, A1, C);
    // QKV GEMM (M x 1152, K=384)
    gemm_wmma<0><<<dim3(M / 128, QN / 128), 256, 0, stream>>>(
        A1, wt_qkv, qkv_b, nullptr, qkv, M, QN, C);
    // windowed attention (2048 windows x 12 heads)
    attn_kernel<<<2048 * 12, 64, 0, stream>>>(qkv, abias, bidx, attn_o, noff);
    // proj GEMM + residual(x) -> x1
    gemm_wmma<1><<<dim3(M / 128, C / 128), 256, 0, stream>>>(
        attn_o, wt_proj, proj_b, x, x1, M, C, C);
    // depthwise conv + BN -> x2 (replaces x per reference)
    dwconv_bn<<<((size_t)M * C) / 256, 256, 0, stream>>>(
        x1, conv_w, bn_w, bn_b, bn_mean, bn_var, x2);
    // LN2 -> f16
    ln_f16_kernel<<<M, 128, 0, stream>>>(x2, ln2_g, ln2_b, A3, C);
    // FC1 GEMM + GELU -> f16
    gemm_wmma<2><<<dim3(M / 128, HID / 128), 256, 0, stream>>>(
        A3, wt_fc1, fc1_b, nullptr, hmid, M, HID, C);
    // FC2 GEMM + residual(x2) -> out
    gemm_wmma<1><<<dim3(M / 128, C / 128), 256, 0, stream>>>(
        hmid, wt_fc2, fc2_b, x2, (float*)d_out, M, C, HID);
}
